// SpatialGraphContrastiveLoss_41480794144812
// MI455X (gfx1250) — compile-verified, matrix operand order checked by reading, not run on hardware
//
#include <hip/hip_runtime.h>

typedef _Float16 f16;
typedef __attribute__((ext_vector_type(4)))  _Float16 v4h;
typedef __attribute__((ext_vector_type(8)))  _Float16 v8h;
typedef __attribute__((ext_vector_type(16))) _Float16 v16h;
typedef __attribute__((ext_vector_type(2)))  float    v2f;
typedef __attribute__((ext_vector_type(4)))  float    v4f;
typedef __attribute__((ext_vector_type(8)))  float    v8f;

#define DIM        128
#define TAU_INV    10.0f
#define LOGIT_MAX  10.0f            // cosine <= 1  =>  logit <= 1/tau
#define L2E_TAU    14.4269504089f   // (1/tau) * log2(e): exp(logit-10)=2^(cs*K-K)
#define EPS2       0.25f            // dist < 0.5  <=>  d2 < 0.25

// ---------------------------------------------------------------------------
// Kernel 1: row-normalize h (fp32 -> unit fp16), per-row |pos|^2, zero accums.
// ---------------------------------------------------------------------------
__global__ void __launch_bounds__(256)
normalize_kernel(const float* __restrict__ h, const float* __restrict__ pos,
                 f16* __restrict__ h16, float* __restrict__ sq,
                 float* __restrict__ acc, int N) {
    if (blockIdx.x == 0 && threadIdx.x < 2) acc[threadIdx.x] = 0.0f;

    const int wave = threadIdx.x >> 5;
    const int lane = threadIdx.x & 31;
    const int row  = blockIdx.x * 8 + wave;
    if (row >= N) return;

    const float* hr = h + (size_t)row * DIM;
    v4f v = *(const v4f*)(hr + lane * 4);
    float ss = v.x * v.x + v.y * v.y + v.z * v.z + v.w * v.w;
    #pragma unroll
    for (int off = 1; off < 32; off <<= 1) ss += __shfl_xor(ss, off, 32);

    const float inv = 1.0f / fmaxf(sqrtf(ss), 1e-12f);
    v4h o;
    o.x = (f16)(v.x * inv); o.y = (f16)(v.y * inv);
    o.z = (f16)(v.z * inv); o.w = (f16)(v.w * inv);
    *(v4h*)(h16 + (size_t)row * DIM + lane * 4) = o;

    if (lane == 0) {
        const float px = pos[row * 3 + 0];
        const float py = pos[row * 3 + 1];
        const float pz = pos[row * 3 + 2];
        sq[row] = px * px + py * py + pz * pz;
    }
}

// ---------------------------------------------------------------------------
// Branchless f32-WMMA position fragment: lanes 0-15 carry (x,y) = K 0,1;
// lanes 16-31 carry (z,0) = K 2,3.
// ---------------------------------------------------------------------------
__device__ __forceinline__ v2f pos_frag(const float* __restrict__ pos, int idx, int hi) {
    const float px = pos[idx * 3 + 0];
    const float py = pos[idx * 3 + 1];
    const float pz = pos[idx * 3 + 2];
    v2f f;
    f.x = hi ? pz   : px;
    f.y = hi ? 0.0f : py;
    return f;
}

// ---------------------------------------------------------------------------
// Operand bundle for one pair of column tiles.
// ---------------------------------------------------------------------------
struct TilePair {
    v16h  B0[4], B1[4];
    v2f   bp0, bp1;
    float sj0, sj1;
    int   j0, j1;
};

__device__ __forceinline__ void load_pair(const f16* __restrict__ h16,
                                          const float* __restrict__ pos,
                                          const float* __restrict__ sq,
                                          int jt, int ln, int hi, TilePair& T) {
    T.j0 = jt * 16 + ln;
    T.j1 = T.j0 + 16;
    const f16* br0 = h16 + (size_t)T.j0 * DIM;
    const f16* br1 = h16 + (size_t)T.j1 * DIM;
    #pragma unroll
    for (int s = 0; s < 4; ++s) {
        v8h l0 = *(const v8h*)(br0 + 32 * s + 8 * hi);
        v8h u0 = *(const v8h*)(br0 + 32 * s + 16 + 8 * hi);
        T.B0[s] = __builtin_shufflevector(l0, u0, 0,1,2,3,4,5,6,7,8,9,10,11,12,13,14,15);
        v8h l1 = *(const v8h*)(br1 + 32 * s + 8 * hi);
        v8h u1 = *(const v8h*)(br1 + 32 * s + 16 + 8 * hi);
        T.B1[s] = __builtin_shufflevector(l1, u1, 0,1,2,3,4,5,6,7,8,9,10,11,12,13,14,15);
    }
    T.bp0 = pos_frag(pos, T.j0, hi);
    T.bp1 = pos_frag(pos, T.j1, hi);
    T.sj0 = sq[T.j0];
    T.sj1 = sq[T.j1];
}

__device__ __forceinline__ void mm_pair(const v16h A[4], const v2f& apos,
                                        const TilePair& T,
                                        v8f& c0, v8f& c1, v8f& d0, v8f& d1) {
    d0 = __builtin_amdgcn_wmma_f32_16x16x4_f32(false, apos, false, T.bp0,
                                               (short)0, v8f{}, false, false);
    d1 = __builtin_amdgcn_wmma_f32_16x16x4_f32(false, apos, false, T.bp1,
                                               (short)0, v8f{}, false, false);
    c0 = v8f{};
    c1 = v8f{};
    #pragma unroll
    for (int s = 0; s < 4; ++s) {
        c0 = __builtin_amdgcn_wmma_f32_16x16x32_f16(false, A[s], false, T.B0[s],
                                                    (short)0, c0, false, false);
        c1 = __builtin_amdgcn_wmma_f32_16x16x32_f16(false, A[s], false, T.B1[s],
                                                    (short)0, c1, false, false);
    }
}

// ---------------------------------------------------------------------------
// Per-tile epilogue, specialized on whether this tile can touch the diagonal.
// P accumulates raw cosines (scaled by 1/tau at the very end).
// ---------------------------------------------------------------------------
template <bool DIAG>
__device__ __forceinline__ void epilogue(const v8f& c, const v8f& d,
                                         int iBase, int hi, int j,
                                         const float* __restrict__ si, float sj,
                                         float* __restrict__ S,
                                         float* __restrict__ P,
                                         float* __restrict__ Cc) {
    #pragma unroll
    for (int r = 0; r < 8; ++r) {
        const float cs = c[r];
        // exp(logit-10) == 2^(cs*K - K), arg in [-29,0] -> raw v_exp_f32 is safe
        float e = __builtin_amdgcn_exp2f(__builtin_fmaf(cs, L2E_TAU, -L2E_TAU));
        // d2 = |pi|^2 + |pj|^2 - 2*pi.pj  (same formulation as reference)
        const float d2 = __builtin_fmaf(d[r], -2.0f, si[r] + sj);
        bool self = false;
        if (DIAG) {
            self = (iBase + r + 8 * hi) == j;
            if (self) e = 0.0f;
        }
        S[r] += e;
        const bool  isp = (d2 < EPS2) && !self;
        const float m   = isp ? 1.0f : 0.0f;
        P[r]  = __builtin_fmaf(m, cs, P[r]);
        Cc[r] += m;
    }
}

template <bool DIAG>
__device__ __forceinline__ void epilogue_pair(const v8f& c0, const v8f& c1,
                                              const v8f& d0, const v8f& d1,
                                              int iBase, int hi, int j0, int j1,
                                              const float* __restrict__ si,
                                              float sj0, float sj1,
                                              float* S, float* P, float* C) {
    epilogue<DIAG>(c0, d0, iBase, hi, j0, si, sj0, S, P, C);
    epilogue<DIAG>(c1, d1, iBase, hi, j1, si, sj1, S, P, C);
}

// ---------------------------------------------------------------------------
// Kernel 2: fused sim-GEMM + loss, software-pipelined: the epilogue of pair p
// executes while the WMMA chains of pair p+1 are in flight, filling the
// WMMA->WMMA hazard slots with independent VALU.
// ---------------------------------------------------------------------------
__global__ void __launch_bounds__(128)
loss_kernel(const f16* __restrict__ h16, const float* __restrict__ pos,
            const float* __restrict__ sq, float* __restrict__ acc, int N) {
    const int lane  = threadIdx.x & 31;
    const int wave  = threadIdx.x >> 5;
    const int hi    = lane >> 4;
    const int ln    = lane & 15;
    const int iBase = blockIdx.x * 16;
    const int myT   = blockIdx.x;
    const int nT    = N >> 4;

    // --- resident A fragments (f16 sim GEMM + f32 position GEMM) ------------
    const f16* arow = h16 + (size_t)(iBase + ln) * DIM;
    v16h A[4];
    #pragma unroll
    for (int s = 0; s < 4; ++s) {
        v8h lo = *(const v8h*)(arow + 32 * s + 8 * hi);
        v8h up = *(const v8h*)(arow + 32 * s + 16 + 8 * hi);
        A[s] = __builtin_shufflevector(lo, up, 0,1,2,3,4,5,6,7,8,9,10,11,12,13,14,15);
    }
    const v2f apos = pos_frag(pos, iBase + ln, hi);

    float si[8];
    #pragma unroll
    for (int r = 0; r < 8; ++r) si[r] = sq[iBase + r + 8 * hi];

    float S[8], P[8], C[8];
    #pragma unroll
    for (int r = 0; r < 8; ++r) { S[r] = 0.0f; P[r] = 0.0f; C[r] = 0.0f; }

    // --- software-pipelined sweep over column-tile pairs --------------------
    TilePair T;
    v8f c0, c1, d0, d1;
    int jprev = wave * 2;
    load_pair(h16, pos, sq, jprev, ln, hi, T);
    mm_pair(A, apos, T, c0, c1, d0, d1);

    for (int jt = jprev + 8; jt < nT; jt += 8) {
        // save prev pair's scalars + accumulator tiles (register renames)
        const int   pj0 = T.j0, pj1 = T.j1;
        const float ps0 = T.sj0, ps1 = T.sj1;
        const v8f pc0 = c0, pc1 = c1, pd0 = d0, pd1 = d1;

        load_pair(h16, pos, sq, jt, ln, hi, T);
        mm_pair(A, apos, T, c0, c1, d0, d1);

        // previous pair's epilogue overlaps the WMMAs above
        if (jprev == myT || jprev + 1 == myT)
            epilogue_pair<true >(pc0, pc1, pd0, pd1, iBase, hi, pj0, pj1, si, ps0, ps1, S, P, C);
        else
            epilogue_pair<false>(pc0, pc1, pd0, pd1, iBase, hi, pj0, pj1, si, ps0, ps1, S, P, C);
        jprev = jt;
    }

    // drain the last pair
    if (jprev == myT || jprev + 1 == myT)
        epilogue_pair<true >(c0, c1, d0, d1, iBase, hi, T.j0, T.j1, si, T.sj0, T.sj1, S, P, C);
    else
        epilogue_pair<false>(c0, c1, d0, d1, iBase, hi, T.j0, T.j1, si, T.sj0, T.sj1, S, P, C);

    // --- reduce the 16 lanes of each half-wave ------------------------------
    #pragma unroll
    for (int r = 0; r < 8; ++r) {
        #pragma unroll
        for (int off = 1; off < 16; off <<= 1) {
            S[r] += __shfl_xor(S[r], off, 32);
            P[r] += __shfl_xor(P[r], off, 32);
            C[r] += __shfl_xor(C[r], off, 32);
        }
    }

    __shared__ float sm[4][16][3];
    if (ln == 0) {
        #pragma unroll
        for (int r = 0; r < 8; ++r) {
            const int row16 = r + 8 * hi;
            sm[wave][row16][0] = S[r];
            sm[wave][row16][1] = P[r];
            sm[wave][row16][2] = C[r];
        }
    }
    __syncthreads();

    // --- merge the 4 waves, emit per-anchor loss ----------------------------
    if (threadIdx.x < 16) {
        const int t = threadIdx.x;
        float Ss = 0.0f, Ps = 0.0f, Cs = 0.0f;
        #pragma unroll
        for (int w = 0; w < 4; ++w) {
            Ss += sm[w][t][0];
            Ps += sm[w][t][1];
            Cs += sm[w][t][2];
        }
        const float lse = logf(Ss) + LOGIT_MAX;
        float loss = 0.0f, valid = 0.0f;
        if (Cs > 0.0f) {
            loss  = -(TAU_INV * Ps - Cs * lse) / Cs;   // P held raw cosines
            valid = 1.0f;
        }
        atomicAdd(&acc[0], loss);
        atomicAdd(&acc[1], valid);
    }
}

// ---------------------------------------------------------------------------
__global__ void finalize_kernel(const float* __restrict__ acc, float* __restrict__ out) {
    out[0] = acc[0] / fmaxf(acc[1], 1.0f);
}

// ---------------------------------------------------------------------------
extern "C" void kernel_launch(void* const* d_in, const int* in_sizes, int n_in,
                              void* d_out, int out_size, void* d_ws, size_t ws_size,
                              hipStream_t stream) {
    (void)n_in; (void)out_size; (void)ws_size;
    const float* h   = (const float*)d_in[0];   // [N,128] fp32
    const float* pos = (const float*)d_in[1];   // [N,3]   fp32
    const int N = in_sizes[0] / DIM;            // 8192

    f16*   h16 = (f16*)d_ws;                                              // 2 MB
    float* sq  = (float*)((char*)d_ws + (size_t)N * DIM * sizeof(f16));   // 32 KB
    float* acc = sq + N;                                                  // 2 floats

    normalize_kernel<<<N / 8, 256, 0, stream>>>(h, pos, h16, sq, acc, N);
    loss_kernel<<<N / 16, 128, 0, stream>>>(h16, pos, sq, acc, N);
    finalize_kernel<<<1, 1, 0, stream>>>(acc, (float*)d_out);
}
